// AttentionBlock_47742856462565
// MI455X (gfx1250) — compile-verified, matrix operand order checked by reading, not run on hardware
//
#include <hip/hip_runtime.h>
#include <hip/hip_bf16.h>
#include <math.h>

typedef __attribute__((ext_vector_type(16))) __bf16 v16bf;
typedef __attribute__((ext_vector_type(8)))  float  v8f;
typedef unsigned short u16;
typedef unsigned int   u32;

constexpr int NB  = 4;
constexpr int NS  = 2048;
constexpr int ND  = 512;
constexpr int NH  = 8;
constexpr int NHD = 64;
constexpr int NM  = NB * NS;   // 8192 token rows

// ---------------------------------------------------------------- helpers ---

__device__ __forceinline__ u16 f2bf(float x) {
  u32 u = __float_as_uint(x);
  u32 r = u + 0x7fffu + ((u >> 16) & 1u);   // round to nearest even
  return (u16)(r >> 16);
}

union FragBF {
  v16bf v;
  u32   u[8];
  uint4 d4[2];
};

// 16x32 bf16 fragment (A layout; also valid as B when operand is (N,K)
// row-major). Per-lane K mapping: k = (lane>>4)*8 + (v&3)*2 + e + (v>>2)*16.
// Two 16-byte loads: elements [p, p+7] and [p+16, p+23].
__device__ __forceinline__ void load_frag(FragBF& f, const u16* base,
                                          int ld, int row, int k0, int lane) {
  const u16* p = base + (size_t)row * ld + k0 + ((lane >> 4) << 3);
  f.d4[0] = *(const uint4*)(p);
  f.d4[1] = *(const uint4*)(p + 16);
}

__device__ __forceinline__ v8f wmma_bf16(const v16bf& a, const v16bf& b, const v8f& c) {
  return __builtin_amdgcn_wmma_f32_16x16x32_bf16(false, a, false, b, (short)0, c,
                                                 false, false);
}

__device__ __forceinline__ v8f zero8() {
  v8f z = {0.f, 0.f, 0.f, 0.f, 0.f, 0.f, 0.f, 0.f};
  return z;
}

// Async global -> LDS copy of 16 bytes per lane (ASYNCcnt-tracked).
__device__ __forceinline__ void async_copy16(u16* lds_ptr, const u16* gptr) {
  u32 lds_off = (u32)(uintptr_t)lds_ptr;   // low 32 bits of flat addr = LDS offset
  asm volatile("global_load_async_to_lds_b128 %0, %1, off"
               :: "v"(lds_off), "v"(gptr)
               : "memory");
}

template <int N>
__device__ __forceinline__ void wait_async() {
  asm volatile("s_wait_asynccnt %0" :: "i"(N) : "memory");
}

// -------------------------------------------------------------- fp32->bf16 --

__global__ void cvt_bf16_k(const float* __restrict__ in, u16* __restrict__ out, int n) {
  int i = blockIdx.x * blockDim.x + threadIdx.x;
  if (i < n) out[i] = f2bf(in[i]);
}

// -------------------------------------------------------------------- GEMM --
// out[m,n] = sum_k A[m,k] * W[n,k] + bias[n]   (A: MxK bf16, W: NxK bf16)
// Block tile 64x128, 8 waves (2x4), wave tile 32x32. A/B k-slabs staged in
// double-buffered LDS via async copies.
// mode 0: QKV scatter (q scaled 1/8, v stored transposed (B,H,HD,S))
// mode 1: GELU -> bf16 out
// mode 2: + residual -> fp32 out
__global__ __launch_bounds__(256)
void gemm_bf16(const u16* __restrict__ A, const u16* __restrict__ W,
               const float* __restrict__ bias,
               int M, int N, int K, int mode,
               float* __restrict__ outf, u16* __restrict__ outh,
               const float* __restrict__ res,
               u16* __restrict__ qb, u16* __restrict__ kb, u16* __restrict__ vtb) {
  __shared__ u16 smA[2][64 * 32];
  __shared__ u16 smB[2][128 * 32];

  const int tid  = threadIdx.x;
  const int lane = tid & 31;
  const int wid  = tid >> 5;
  const int wm = wid & 1, wn = wid >> 1;                  // 2x4 wave grid
  const int mblk = blockIdx.y * 64;
  const int nblk = blockIdx.x * 128;
  const int m0 = wm * 32, n0 = wn * 32;                   // within block tile
  const int lr = lane & 15;

  // copy assignment: 16B chunk per async op; A: 256 chunks, B: 512 chunks
  const int crow   = tid >> 2;                            // 0..63
  const int cchunk = (tid & 3) << 3;                      // element offset 0/8/16/24

  auto issue_tile = [&](int buf, int k0) {
    async_copy16(&smA[buf][crow * 32 + cchunk],
                 A + (size_t)(mblk + crow) * K + k0 + cchunk);
    async_copy16(&smB[buf][crow * 32 + cchunk],
                 W + (size_t)(nblk + crow) * K + k0 + cchunk);
    async_copy16(&smB[buf][(64 + crow) * 32 + cchunk],
                 W + (size_t)(nblk + 64 + crow) * K + k0 + cchunk);
  };

  v8f acc00 = zero8(), acc01 = zero8(), acc10 = zero8(), acc11 = zero8();

  const int nk = K >> 5;
  issue_tile(0, 0);
  for (int i = 0; i < nk; i++) {
    const int cur = i & 1;
    if (i + 1 < nk) {
      issue_tile(cur ^ 1, (i + 1) << 5);
      wait_async<3>();                  // slab i complete, slab i+1 in flight
    } else {
      wait_async<0>();
    }
    __syncthreads();
    FragBF a0, a1, b0, b1;
    load_frag(a0, smA[cur], 32, m0 + lr,      0, lane);
    load_frag(a1, smA[cur], 32, m0 + 16 + lr, 0, lane);
    load_frag(b0, smB[cur], 32, n0 + lr,      0, lane);
    load_frag(b1, smB[cur], 32, n0 + 16 + lr, 0, lane);
    acc00 = wmma_bf16(a0.v, b0.v, acc00);
    acc01 = wmma_bf16(a0.v, b1.v, acc01);
    acc10 = wmma_bf16(a1.v, b0.v, acc10);
    acc11 = wmma_bf16(a1.v, b1.v, acc11);
    __syncthreads();                    // safe to overwrite buf `cur` next iter
  }

  const int rowoff = (lane >> 4) << 3;
  v8f accs[2][2] = {{acc00, acc01}, {acc10, acc11}};
  #pragma unroll
  for (int i = 0; i < 2; i++) {
    #pragma unroll
    for (int j = 0; j < 2; j++) {
      const int nn = nblk + n0 + 16 * j + lr;
      const float bv = bias ? bias[nn] : 0.f;
      #pragma unroll
      for (int r = 0; r < 8; r++) {
        const int mm = mblk + m0 + 16 * i + rowoff + r;
        float v = accs[i][j][r] + bv;
        if (mode == 0) {
          const int which = nn >> 9;            // 0=q 1=k 2=v
          const int dcol  = nn & 511;
          const int h = dcol >> 6, hd = dcol & 63;
          const int bb = mm >> 11, s = mm & (NS - 1);
          const size_t bh = (size_t)(bb * NH + h);
          if (which == 0)      qb [(bh * NS + s) * NHD + hd] = f2bf(v * 0.125f);
          else if (which == 1) kb [(bh * NS + s) * NHD + hd] = f2bf(v);
          else                 vtb[(bh * NHD + hd) * NS + s] = f2bf(v);
        } else if (mode == 1) {
          float g = 0.5f * v * (1.f + erff(v * 0.70710678118654752f));
          outh[(size_t)mm * N + nn] = f2bf(g);
        } else {
          outf[(size_t)mm * N + nn] = v + res[(size_t)mm * N + nn];
        }
      }
    }
  }
}

// --------------------------------------------------------- flash attention --
// grid: (S/64, B*H), 4 waves/block, each wave owns 16 q rows.
// Scores computed transposed (S^T = K * Q^T) so softmax over k is lane-local;
// the S^T accumulator layout IS the A-fragment layout for P*V.
// K-block (64x64) and V^T-block (64x64) are shared by all 4 waves -> staged in
// double-buffered LDS with async copies.
__global__ __launch_bounds__(128)
void flash_attn(const u16* __restrict__ qb, const u16* __restrict__ kbuf,
                const u16* __restrict__ vtb, u16* __restrict__ ctx) {
  __shared__ u16 smK[2][64 * 64];       // [k row][hd]
  __shared__ u16 smV[2][64 * 64];       // [hd][k row]

  const int tid    = threadIdx.x;
  const int lane   = tid & 31;
  const int wid    = tid >> 5;
  const int lr     = lane & 15;
  const int rowoff = (lane >> 4) << 3;
  const int bh = blockIdx.y;
  const int q0 = blockIdx.x * 64 + wid * 16;

  const u16* qB = qb   + (size_t)bh * NS * NHD;
  const u16* kB = kbuf + (size_t)bh * NS * NHD;
  const u16* vB = vtb  + (size_t)bh * NHD * NS;

  // 512 chunks (16B) per array, 128 threads -> 4 chunks each
  auto issue_kv = [&](int buf, int kb0) {
    #pragma unroll
    for (int c = 0; c < 4; c++) {
      const int id  = tid + c * 128;          // 0..511
      const int row = id >> 3;                // 0..63
      const int off = (id & 7) << 3;          // element offset 0..56
      async_copy16(&smK[buf][row * 64 + off], kB + (size_t)(kb0 + row) * NHD + off);
      async_copy16(&smV[buf][row * 64 + off], vB + (size_t)row * NS + kb0 + off);
    }
  };

  FragBF qf0, qf1;                            // B fragments (n = q, k = hd)
  load_frag(qf0, qB, NHD, q0 + lr, 0,  lane);
  load_frag(qf1, qB, NHD, q0 + lr, 32, lane);

  float mrun = -1e30f, lrun = 0.f;
  v8f cacc[4] = {zero8(), zero8(), zero8(), zero8()};

  issue_kv(0, 0);
  for (int kb0 = 0; kb0 < NS; kb0 += 64) {
    const int cur = (kb0 >> 6) & 1;
    if (kb0 + 64 < NS) {
      issue_kv(cur ^ 1, kb0 + 64);
      wait_async<8>();                  // block kb0 complete, next in flight
    } else {
      wait_async<0>();
    }
    __syncthreads();

    // ---- S^T tiles: rows = k (4 tiles of 16), cols = q ----
    v8f st[4];
    #pragma unroll
    for (int j = 0; j < 4; j++) {
      FragBF ka0, ka1;
      load_frag(ka0, smK[cur], 64, 16 * j + lr, 0,  lane);
      load_frag(ka1, smK[cur], 64, 16 * j + lr, 32, lane);
      st[j] = wmma_bf16(ka0.v, qf0.v, zero8());
      st[j] = wmma_bf16(ka1.v, qf1.v, st[j]);
    }
    // ---- online softmax (per q = lane&15; halves combine via xor 16) ----
    float bmax = -1e30f;
    #pragma unroll
    for (int j = 0; j < 4; j++)
      #pragma unroll
      for (int r = 0; r < 8; r++) bmax = fmaxf(bmax, st[j][r]);
    bmax = fmaxf(bmax, __shfl_xor(bmax, 16, 32));
    const float mn = fmaxf(mrun, bmax);
    const float alpha = __expf(mrun - mn);
    float bsum = 0.f;
    #pragma unroll
    for (int j = 0; j < 4; j++)
      #pragma unroll
      for (int r = 0; r < 8; r++) {
        float p = __expf(st[j][r] - mn);
        st[j][r] = p;
        bsum += p;
      }
    bsum += __shfl_xor(bsum, 16, 32);
    lrun = lrun * alpha + bsum;
    mrun = mn;
    // rescale ctx accumulators (q lives on the M axis there -> broadcast)
    float al[8];
    #pragma unroll
    for (int r = 0; r < 8; r++) al[r] = __shfl(alpha, rowoff + r, 32);
    #pragma unroll
    for (int t = 0; t < 4; t++)
      #pragma unroll
      for (int r = 0; r < 8; r++) cacc[t][r] *= al[r];
    // ---- repack P (registers) into A fragments for P*V ----
    FragBF pf[2];
    #pragma unroll
    for (int c = 0; c < 2; c++) {
      #pragma unroll
      for (int vv = 0; vv < 4; vv++)
        pf[c].u[vv] = (u32)f2bf(st[2 * c][2 * vv]) |
                      ((u32)f2bf(st[2 * c][2 * vv + 1]) << 16);
      #pragma unroll
      for (int vv = 4; vv < 8; vv++)
        pf[c].u[vv] = (u32)f2bf(st[2 * c + 1][2 * (vv - 4)]) |
                      ((u32)f2bf(st[2 * c + 1][2 * (vv - 4) + 1]) << 16);
    }
    // ---- ctx += P * V  (V^T in LDS: B frag n = hd, k = key idx) ----
    #pragma unroll
    for (int t = 0; t < 4; t++) {
      FragBF vf0, vf1;
      load_frag(vf0, smV[cur], 64, 16 * t + lr, 0,  lane);
      load_frag(vf1, smV[cur], 64, 16 * t + lr, 32, lane);
      cacc[t] = wmma_bf16(pf[0].v, vf0.v, cacc[t]);
      cacc[t] = wmma_bf16(pf[1].v, vf1.v, cacc[t]);
    }
    __syncthreads();
  }

  float linv[8];
  #pragma unroll
  for (int r = 0; r < 8; r++) linv[r] = 1.f / __shfl(lrun, rowoff + r, 32);
  const int b = bh / NH, h = bh % NH;
  #pragma unroll
  for (int t = 0; t < 4; t++)
    #pragma unroll
    for (int r = 0; r < 8; r++) {
      const int qg = q0 + rowoff + r;
      const int hd = 16 * t + lr;
      ctx[((size_t)b * NS + qg) * ND + h * NHD + hd] = f2bf(cacc[t][r] * linv[r]);
    }
}

// --------------------------------------------------------------- layernorm --

__global__ __launch_bounds__(256)
void layernorm_k(const float* __restrict__ in, const float* __restrict__ g,
                 const float* __restrict__ bta, float* __restrict__ y,
                 u16* __restrict__ yb) {
  const int row = blockIdx.x;
  const float* xr = in + (size_t)row * ND;
  const int t = threadIdx.x;
  const float v0 = xr[t], v1 = xr[t + 256];
  float s = v0 + v1, s2 = v0 * v0 + v1 * v1;
  #pragma unroll
  for (int o = 16; o > 0; o >>= 1) {
    s  += __shfl_xor(s,  o, 32);
    s2 += __shfl_xor(s2, o, 32);
  }
  __shared__ float ssum[8], ssum2[8];
  const int wid = t >> 5, lane = t & 31;
  if (lane == 0) { ssum[wid] = s; ssum2[wid] = s2; }
  __syncthreads();
  if (wid == 0) {
    float a  = lane < 8 ? ssum[lane]  : 0.f;
    float a2 = lane < 8 ? ssum2[lane] : 0.f;
    #pragma unroll
    for (int o = 4; o > 0; o >>= 1) {
      a  += __shfl_xor(a,  o, 32);
      a2 += __shfl_xor(a2, o, 32);
    }
    if (lane == 0) { ssum[0] = a; ssum2[0] = a2; }
  }
  __syncthreads();
  const float mean = ssum[0] * (1.f / ND);
  const float var  = ssum2[0] * (1.f / ND) - mean * mean;
  const float rstd = rsqrtf(var + 1e-5f);
  const float o0 = (v0 - mean) * rstd * g[t]       + bta[t];
  const float o1 = (v1 - mean) * rstd * g[t + 256] + bta[t + 256];
  y[(size_t)row * ND + t]       = o0;
  y[(size_t)row * ND + t + 256] = o1;
  if (yb) {
    yb[(size_t)row * ND + t]       = f2bf(o0);
    yb[(size_t)row * ND + t + 256] = f2bf(o1);
  }
}

// ------------------------------------------------------------------ launch --

extern "C" void kernel_launch(void* const* d_in, const int* in_sizes, int n_in,
                              void* d_out, int out_size, void* d_ws, size_t ws_size,
                              hipStream_t stream) {
  (void)in_sizes; (void)n_in; (void)out_size; (void)ws_size;
  const float* x    = (const float*)d_in[0];
  const float* wqkv = (const float*)d_in[1];
  const float* bqkv = (const float*)d_in[2];
  const float* wo   = (const float*)d_in[3];
  const float* bo   = (const float*)d_in[4];
  const float* g1   = (const float*)d_in[5];
  const float* bt1  = (const float*)d_in[6];
  const float* g2   = (const float*)d_in[7];
  const float* bt2  = (const float*)d_in[8];
  const float* w1   = (const float*)d_in[9];
  const float* bb1  = (const float*)d_in[10];
  const float* w2   = (const float*)d_in[11];
  const float* bb2  = (const float*)d_in[12];
  float* out = (float*)d_out;

  char* ws = (char*)d_ws;
  auto alloc = [&](size_t bytes) {
    char* p = ws;
    ws += (bytes + 255) & ~(size_t)255;
    return p;
  };
  u16*   xb    = (u16*)  alloc((size_t)NM * ND * 2);
  u16*   wqkvb = (u16*)  alloc((size_t)3 * ND * ND * 2);
  u16*   wob   = (u16*)  alloc((size_t)ND * ND * 2);
  u16*   w1b   = (u16*)  alloc((size_t)4 * ND * ND * 2);
  u16*   w2b   = (u16*)  alloc((size_t)ND * 4 * ND * 2);
  u16*   qbuf  = (u16*)  alloc((size_t)NM * ND * 2);
  u16*   kbuf  = (u16*)  alloc((size_t)NM * ND * 2);
  u16*   vtbuf = (u16*)  alloc((size_t)NM * ND * 2);
  u16*   ctxb  = (u16*)  alloc((size_t)NM * ND * 2);
  float* res1  = (float*)alloc((size_t)NM * ND * 4);
  float* y1    = (float*)alloc((size_t)NM * ND * 4);
  u16*   y1b   = (u16*)  alloc((size_t)NM * ND * 2);
  u16*   hb    = (u16*)  alloc((size_t)NM * 4 * ND * 2);
  float* res2  = (float*)alloc((size_t)NM * ND * 4);

  auto cvt = [&](const float* src, u16* dst, int n) {
    cvt_bf16_k<<<(n + 255) / 256, 256, 0, stream>>>(src, dst, n);
  };
  cvt(x,    xb,    NM * ND);
  cvt(wqkv, wqkvb, 3 * ND * ND);
  cvt(wo,   wob,   ND * ND);
  cvt(w1,   w1b,   4 * ND * ND);
  cvt(w2,   w2b,   ND * 4 * ND);

  // QKV projection + head scatter (q scaled, v transposed)
  gemm_bf16<<<dim3((3 * ND) / 128, NM / 64), 256, 0, stream>>>(
      xb, wqkvb, bqkv, NM, 3 * ND, ND, 0,
      nullptr, nullptr, nullptr, qbuf, kbuf, vtbuf);

  // flash attention
  flash_attn<<<dim3(NS / 64, NB * NH), 128, 0, stream>>>(qbuf, kbuf, vtbuf, ctxb);

  // out projection + residual
  gemm_bf16<<<dim3(ND / 128, NM / 64), 256, 0, stream>>>(
      ctxb, wob, bo, NM, ND, ND, 2,
      res1, nullptr, x, nullptr, nullptr, nullptr);

  layernorm_k<<<NM, 256, 0, stream>>>(res1, g1, bt1, y1, y1b);

  // FFN up + GELU
  gemm_bf16<<<dim3((4 * ND) / 128, NM / 64), 256, 0, stream>>>(
      y1b, w1b, bb1, NM, 4 * ND, ND, 1,
      nullptr, hb, nullptr, nullptr, nullptr, nullptr);

  // FFN down + residual
  gemm_bf16<<<dim3(ND / 128, NM / 64), 256, 0, stream>>>(
      hb, w2b, bb2, NM, ND, 4 * ND, 2,
      res2, nullptr, y1, nullptr, nullptr, nullptr);

  layernorm_k<<<NM, 256, 0, stream>>>(res2, g2, bt2, out, nullptr);
}
